// DifferentiableNeuralGas_61469571940956
// MI455X (gfx1250) — matrix operand order repaired and checked by reading, not compiled, not run on hardware
//
#include <hip/hip_runtime.h>
#include <stdint.h>

typedef __attribute__((ext_vector_type(16))) _Float16 v16h;
typedef __attribute__((ext_vector_type(8)))  float    v8f;

#define N_ROWS   16384
#define K_CENT   128
#define DIM      64
#define TILE_M   16
#define TAU_INV    5.0f                         // 1/0.2
#define LOG2E      1.44269504088896340736f
#define SIG_SCALE  (TAU_INV * LOG2E)            // exp(-x/tau) == exp2(-x*SIG_SCALE)
#define NB_SCALE   (0.125f * LOG2E)             // exp(-x/8)   == exp2(-x*NB_SCALE)

// Kernel 1: per-workgroup tile of 16 data rows.
//   Phase A: async-DMA the f32 data tile into LDS (global_load_async_to_lds_b128),
//            convert weights to f16 in LDS while it flies, f32 row norms.
//   Phase B: 8 waves x 2x v_wmma_f32_16x16x32_f16 -> 16x128 dot tile -> distances.
//   Phase C: K x K soft-rank sigmoids (v_exp_f32 / v_rcp_f32), neighborhood
//            weighting, block reduction -> one partial sum per workgroup.
__global__ __launch_bounds__(256) void dng_main(const float* __restrict__ data,
                                                const float* __restrict__ weights,
                                                float* __restrict__ partial)
{
    __shared__ float    sXf[TILE_M * DIM];        //  4 KB raw f32 data tile (async-staged)
    __shared__ _Float16 sW[K_CENT * DIM];         // 16 KB weights as f16 [K][D]
    __shared__ _Float16 sX[TILE_M * DIM];         //  2 KB data tile as f16 [16][D]
    __shared__ float    sDist[TILE_M * K_CENT];   //  8 KB distance tile
    __shared__ float    sW2[K_CENT];
    __shared__ float    sX2[TILE_M];
    __shared__ float    sRed[256];

    const int t       = threadIdx.x;
    const int lane    = t & 31;
    const int wv      = t >> 5;          // wave id 0..7 -> centroid column tile
    const int rowBase = blockIdx.x * TILE_M;

    // ---- Phase A0: async DMA data tile global -> LDS, 16 B per lane (4 KB total) ----
    {
        const float* srcBase = data + rowBase * DIM;          // uniform -> SGPR pair
        uint32_t ldsAddr = (uint32_t)(size_t)&sXf[t * 4];     // LDS byte address
        uint32_t vOff    = (uint32_t)(t * 16);                // per-lane byte offset
        asm volatile("global_load_async_to_lds_b128 %0, %1, %2"
                     :: "v"(ldsAddr), "v"(vOff), "s"(srcBase) : "memory");
    }

    // Prefetch the weight block (gfx1250 global_prefetch_b8 path).
    __builtin_prefetch(&weights[t * 32], 0, 0);

    // ---- Phase A1: weights -> f16 LDS + |w|^2, overlapped with the async DMA ----
    for (int i = t; i < K_CENT * DIM; i += 256) sW[i] = (_Float16)weights[i];
    if (t < K_CENT) {
        float s = 0.0f;
        #pragma unroll 8
        for (int d = 0; d < DIM; ++d) { float v = weights[t * DIM + d]; s += v * v; }
        sW2[t] = s;
    }

    // Wait for this wave's async loads; barrier makes all waves' tiles visible.
    asm volatile("s_wait_asynccnt 0x0" ::: "memory");
    __syncthreads();

    // ---- Phase A2: data tile f16 copy + |x|^2 from the LDS-staged f32 tile ----
    for (int i = t; i < TILE_M * DIM; i += 256) sX[i] = (_Float16)sXf[i];
    if (t < TILE_M) {
        float s = 0.0f;
        #pragma unroll 8
        for (int d = 0; d < DIM; ++d) { float v = sXf[t * DIM + d]; s += v * v; }
        sX2[t] = s;
    }
    __syncthreads();

    // ---- Phase B: WMMA dot products, then distances ----
    // Layouts used here (wave32; A/B take k contiguously, not in AMD's table order):
    //  A 16x32 f16: lane l: M = l&15, g = l>>4; element j (0..15): K = 16g + j
    //  B 32x16 f16: lane l: N = l&15, g = l>>4; element j (0..15): K = 16g + j
    //  C/D 16x16 f32: vgpr r: M = r + 8g, N = lane&15
    const int g   = lane >> 4;
    const int mn  = lane & 15;
    const int col = wv * 16 + mn;        // centroid index for B/C columns

    v16h a0, a1, b0, b1;
    #pragma unroll
    for (int j = 0; j < 16; ++j) {
        const int ka = 16 * g + j;
        a0[j] = sX[mn * DIM + ka];
        a1[j] = sX[mn * DIM + 32 + ka];
        const int kb = 16 * g + j;
        b0[j] = sW[col * DIM + kb];          // B = W^T : B[k][n] = weights[col][k]
        b1[j] = sW[col * DIM + 32 + kb];
    }

    v8f acc = {};
    acc = __builtin_amdgcn_wmma_f32_16x16x32_f16(false, a0, false, b0,
                                                 (short)0, acc, false, false);
    acc = __builtin_amdgcn_wmma_f32_16x16x32_f16(false, a1, false, b1,
                                                 (short)0, acc, false, false);

    #pragma unroll
    for (int r = 0; r < 8; ++r) {
        const int m = r + 8 * g;
        const float d2 = sX2[m] + sW2[col] - 2.0f * acc[r];
        sDist[m * K_CENT + col] = __builtin_amdgcn_sqrtf(fmaxf(d2, 0.0f)); // raw v_sqrt_f32
    }
    __syncthreads();

    // ---- Phase C: soft rank + neighborhood, partial loss ----
    float lsum = 0.0f;
    #pragma unroll
    for (int it = 0; it < (TILE_M * K_CENT) / 256; ++it) {   // 8 (m,k) pairs / thread
        const int idx = t + it * 256;
        const int m   = idx >> 7;                            // idx / 128
        const float dk = sDist[idx];
        const float* drow = &sDist[m * K_CENT];
        float srow = 0.0f;
        #pragma unroll 8
        for (int j = 0; j < K_CENT; ++j) {
            // sigmoid((dk-dj)/tau) = 1 / (1 + exp2((dj-dk)*SIG_SCALE))
            const float e = __builtin_amdgcn_exp2f((drow[j] - dk) * SIG_SCALE);
            srow += __builtin_amdgcn_rcpf(1.0f + e);
        }
        // full-row sum includes j==k contribution sigmoid(0)=0.5; remove it.
        const float sr_m1 = srow - 0.5f;                     // soft_rank - 1
        const float nb = __builtin_amdgcn_exp2f(-sr_m1 * NB_SCALE);
        lsum += nb * dk;
    }

    // ---- block reduction -> one partial per workgroup ----
    sRed[t] = lsum;
    __syncthreads();
    for (int s = 128; s > 0; s >>= 1) {
        if (t < s) sRed[t] += sRed[t + s];
        __syncthreads();
    }
    if (t == 0) partial[blockIdx.x] = sRed[0];
}

// Kernel 2: deterministic reduction of the 1024 workgroup partials -> mean.
__global__ __launch_bounds__(256) void dng_reduce(const float* __restrict__ partial,
                                                  float* __restrict__ out, int n)
{
    __shared__ float sRed[256];
    const int t = threadIdx.x;
    float s = 0.0f;
    for (int i = t; i < n; i += 256) s += partial[i];
    sRed[t] = s;
    __syncthreads();
    for (int k = 128; k > 0; k >>= 1) {
        if (t < k) sRed[t] += sRed[t + k];
        __syncthreads();
    }
    if (t == 0) out[0] = sRed[0] * (1.0f / ((float)N_ROWS * (float)K_CENT));
}

extern "C" void kernel_launch(void* const* d_in, const int* in_sizes, int n_in,
                              void* d_out, int out_size, void* d_ws, size_t ws_size,
                              hipStream_t stream)
{
    (void)in_sizes; (void)n_in; (void)out_size; (void)ws_size;
    const float* data    = (const float*)d_in[0];   // [16384, 64] f32
    const float* weights = (const float*)d_in[1];   // [128, 64]  f32
    float* partial = (float*)d_ws;                  // 1024 floats of scratch
    const int nblocks = N_ROWS / TILE_M;            // 1024 workgroups

    dng_main<<<nblocks, 256, 0, stream>>>(data, weights, partial);
    dng_reduce<<<1, 256, 0, stream>>>(partial, (float*)d_out, nblocks);
}
